// GraphEmbedding_5875515261149
// MI455X (gfx1250) — matrix operand order repaired and checked by reading, not compiled, not run on hardware
//
#include <hip/hip_runtime.h>
#include <hip/hip_bf16.h>

#define N_NODES  50000
#define N_EDGES  400000
#define DIM      512
#define N_LAYERS 5
#define N_GRAPHS 512
#define IN_FEAT  11
#define NPAD     50016   // 1563 * 32  (M padded to WMMA tile multiple)

typedef __attribute__((ext_vector_type(16))) __bf16 v16bf;
typedef __attribute__((ext_vector_type(8)))  __bf16 v8bf;
typedef __attribute__((ext_vector_type(8)))  float  v8f;

__device__ __forceinline__ unsigned short f2bf(float f) {
    union { float f; unsigned u; } v; v.f = f;
    unsigned r = v.u + 0x7FFFu + ((v.u >> 16) & 1u);   // round-to-nearest-even
    return (unsigned short)(r >> 16);
}

// ---------------- init / degree ----------------
__global__ void k_zero_out(float* out) {
    int i = blockIdx.x * blockDim.x + threadIdx.x;
    if (i < N_GRAPHS * DIM) out[i] = 0.0f;
}
__global__ void k_deg_init(float* dinv) {
    int i = blockIdx.x * blockDim.x + threadIdx.x;
    if (i < NPAD) dinv[i] = 2.0f;                       // self-loop weight 2
}
__global__ void k_deg_count(const int* __restrict__ ei, float* dinv) {
    int e = blockIdx.x * blockDim.x + threadIdx.x;
    if (e < N_EDGES) atomicAdd(&dinv[ei[N_EDGES + e]], 1.0f);   // dst = ei[1][e]
}
__global__ void k_deg_fin(float* dinv) {
    int i = blockIdx.x * blockDim.x + threadIdx.x;
    if (i < NPAD) dinv[i] = rsqrtf(dinv[i]);
}

// ---------------- weight packing into WMMA B-fragment layout ----------------
// B frag (16-bit, 32x16): lanes 0-15 = col N=lane, K 0..7/16..23 packed 2/VGPR;
// lanes 16-31 = same cols, K 8..15/24..31. Pack so each lane reads 16 contiguous bf16.
__global__ void k_pack_w(const float* __restrict__ Wg, unsigned short* __restrict__ Wp) {
    int idx = blockIdx.x * blockDim.x + threadIdx.x;    // N_LAYERS*16*32*32*16
    if (idx >= N_LAYERS * 16 * 32 * 32 * 16) return;
    int i    = idx & 15;          // element within lane fragment -> K sub-index
    int lane = (idx >> 4) & 31;
    int nt   = (idx >> 9) & 31;   // 16-wide N tile
    int kt   = (idx >> 14) & 15;  // 32-wide K tile
    int l    = idx >> 18;
    int k = kt * 32 + ((i >> 3) << 4) + ((lane >> 4) << 3) + (i & 7);
    int n = nt * 16 + (lane & 15);
    Wp[idx] = f2bf(Wg[((size_t)l * DIM + k) * DIM + n]);
}

// ---------------- input expansion: h = log(x+1) @ We + be ----------------
__global__ __launch_bounds__(256) void k_embed(const float* __restrict__ x,
                                               const float* __restrict__ We,
                                               const float* __restrict__ be,
                                               float* __restrict__ h,
                                               unsigned short* __restrict__ hb) {
    __shared__ float lx[IN_FEAT];
    const int row = blockIdx.x;
    if (row < N_NODES) {
        if (threadIdx.x < IN_FEAT)
            lx[threadIdx.x] = logf(x[row * IN_FEAT + threadIdx.x] + 1.0f);
        __syncthreads();
        for (int d = threadIdx.x; d < DIM; d += blockDim.x) {
            float acc = be[d];
#pragma unroll
            for (int k = 0; k < IN_FEAT; ++k) acc = fmaf(lx[k], We[k * DIM + d], acc);
            size_t o = (size_t)row * DIM + d;
            h[o]  = acc;
            hb[o] = f2bf(acc);
        }
    } else {  // pad rows: keep bf16 activations zero so WMMA A-loads are defined
        for (int d = threadIdx.x; d < DIM; d += blockDim.x)
            hb[(size_t)row * DIM + d] = 0;
    }
}

// ---------------- hW = h @ Wg[l]  (bf16 WMMA, f32 accum) ----------------
// One wave32 per 32(M) x 64(N) output tile: 2x4 register blocking of
// v_wmma_f32_16x16x32_bf16 -> 8 WMMAs per 12 b128 loads per K-step.
// Stores are UNGUARDED: hW has NPAD rows and pad-row A inputs are zero,
// so pad accumulators are 0 and never read downstream.
__global__ __launch_bounds__(256) void k_gemm(const unsigned short* __restrict__ hb,
                                              const unsigned short* __restrict__ Wp,
                                              float* __restrict__ hW) {
    const int lane = threadIdx.x & 31;
    const int wave = blockIdx.x * 8 + (threadIdx.x >> 5);
    const int mt = wave >> 3;            // 0..1562 (32-row tiles)
    const int nt = wave & 7;             // 0..7    (64-col tiles)
    const int m0 = mt * 32, n0 = nt * 64;
    const int half = lane >> 4;          // A: 0 -> K 0..7/16..23, 1 -> K 8..15/24..31
    const int lid  = lane & 15;

    v8f z = {0.f,0.f,0.f,0.f,0.f,0.f,0.f,0.f};
    v8f acc[2][4] = {{z, z, z, z}, {z, z, z, z}};

    for (int kk = 0; kk < 16; ++kk) {
        const int k0 = kk * 32;
        v16bf afrag[2], bfrag[4];
#pragma unroll
        for (int mi = 0; mi < 2; ++mi) {   // A: 16-bit 16x32 layout (ISA 7.12.2)
            const unsigned short* ap =
                hb + (size_t)(m0 + mi * 16 + lid) * DIM + k0 + half * 8;
            union { v16bf v; v8bf hpart[2]; } ua;
            ua.hpart[0] = *(const v8bf*)(ap);
            ua.hpart[1] = *(const v8bf*)(ap + 16);
            afrag[mi] = ua.v;
        }
#pragma unroll
        for (int ni = 0; ni < 4; ++ni) {   // B: pre-packed contiguous per lane
            const int ntile = (nt << 2) + ni;   // global 16-wide N tile index
            const unsigned short* bp =
                Wp + ((((size_t)kk * 32 + ntile) * 32 + lane) << 4);
            bfrag[ni] = *(const v16bf*)bp;
        }
#pragma unroll
        for (int mi = 0; mi < 2; ++mi)
#pragma unroll
            for (int ni = 0; ni < 4; ++ni)
                acc[mi][ni] = __builtin_amdgcn_wmma_f32_16x16x32_bf16(
                    false, afrag[mi], false, bfrag[ni],
                    (short)0, acc[mi][ni], false, false);
    }

    // C/D layout: lane<16 -> M = j, lane>=16 -> M = j+8; col = lid
#pragma unroll
    for (int mi = 0; mi < 2; ++mi)
#pragma unroll
        for (int ni = 0; ni < 4; ++ni) {
            float* dst = hW + (size_t)(m0 + mi * 16 + half * 8) * DIM
                            + n0 + ni * 16 + lid;
#pragma unroll
            for (int j = 0; j < 8; ++j)
                dst[(size_t)j * DIM] = acc[mi][ni][j];
        }
}

// ---------------- aggregation ----------------
__global__ void k_selfloop(const float* __restrict__ hW, const float* __restrict__ dinv,
                           const float* __restrict__ bg, float* __restrict__ agg) {
    int idx = blockIdx.x * blockDim.x + threadIdx.x;   // N_NODES*DIM
    int row = idx >> 9;
    if (row >= N_NODES) return;
    float s = dinv[row];
    agg[idx] = 2.0f * s * s * hW[idx] + bg[idx & (DIM - 1)];
}

__global__ void k_edges(const int* __restrict__ ei, const float* __restrict__ dinv,
                        const float* __restrict__ hW, float* __restrict__ agg) {
    int idx = blockIdx.x * blockDim.x + threadIdx.x;   // N_EDGES * 128
    int e = idx >> 7;
    if (e >= N_EDGES) return;
    int d = (idx & 127) << 2;
    int s = ei[e], t = ei[N_EDGES + e];
    float norm = dinv[s] * dinv[t];
    const float4 v = *(const float4*)(hW + (size_t)s * DIM + d);
    float* a = agg + (size_t)t * DIM + d;
    atomicAdd(a + 0, norm * v.x);
    atomicAdd(a + 1, norm * v.y);
    atomicAdd(a + 2, norm * v.z);
    atomicAdd(a + 3, norm * v.w);
}

__global__ void k_update(const float* __restrict__ agg, float* __restrict__ h,
                         unsigned short* __restrict__ hb) {
    int idx = blockIdx.x * blockDim.x + threadIdx.x;   // N_NODES*DIM
    if (idx >= N_NODES * DIM) return;
    float v = h[idx] + fmaxf(agg[idx], 0.0f);          // residual + ReLU
    h[idx]  = v;
    hb[idx] = f2bf(v);
}

// ---------------- global_add_pool ----------------
__global__ void k_pool(const float* __restrict__ h, const int* __restrict__ batch,
                       float* __restrict__ out) {
    int idx = blockIdx.x * blockDim.x + threadIdx.x;   // N_NODES * 128
    int row = idx >> 7;
    if (row >= N_NODES) return;
    int d = (idx & 127) << 2;
    int g = batch[row];
    const float4 v = *(const float4*)(h + (size_t)row * DIM + d);
    float* o = out + (size_t)g * DIM + d;
    atomicAdd(o + 0, v.x); atomicAdd(o + 1, v.y);
    atomicAdd(o + 2, v.z); atomicAdd(o + 3, v.w);
}

extern "C" void kernel_launch(void* const* d_in, const int* in_sizes, int n_in,
                              void* d_out, int out_size, void* d_ws, size_t ws_size,
                              hipStream_t stream) {
    const float* x     = (const float*)d_in[0];
    const int*   ei    = (const int*)d_in[1];
    const int*   batch = (const int*)d_in[2];
    /* d_in[3] = batch_size scalar (compile-time constant here) */
    const float* We    = (const float*)d_in[4];
    const float* be    = (const float*)d_in[5];
    const float* Wg    = (const float*)d_in[6];
    const float* bg    = (const float*)d_in[7];
    float* out = (float*)d_out;
    (void)in_sizes; (void)n_in; (void)out_size; (void)ws_size;

    char* ws = (char*)d_ws;
    size_t off = 0;
    auto take = [&](size_t bytes) -> char* {
        char* p = ws + off;
        off = (off + bytes + 255) & ~(size_t)255;
        return p;
    };
    float*          dinv = (float*)take((size_t)NPAD * 4);
    float*          h    = (float*)take((size_t)NPAD * DIM * 4);
    float*          hW   = (float*)take((size_t)NPAD * DIM * 4);
    float*          agg  = (float*)take((size_t)NPAD * DIM * 4);
    unsigned short* hb   = (unsigned short*)take((size_t)NPAD * DIM * 2);
    unsigned short* Wp   = (unsigned short*)take((size_t)N_LAYERS * DIM * DIM * 2);

    k_zero_out <<<(N_GRAPHS * DIM) / 256, 256, 0, stream>>>(out);
    k_deg_init <<<(NPAD + 255) / 256, 256, 0, stream>>>(dinv);
    k_deg_count<<<(N_EDGES + 255) / 256, 256, 0, stream>>>(ei, dinv);
    k_deg_fin  <<<(NPAD + 255) / 256, 256, 0, stream>>>(dinv);
    k_pack_w   <<<(N_LAYERS * DIM * DIM) / 256, 256, 0, stream>>>(Wg, Wp);
    k_embed    <<<NPAD, 256, 0, stream>>>(x, We, be, h, hb);

    for (int l = 0; l < N_LAYERS; ++l) {
        k_gemm    <<<(NPAD / 32) * (DIM / 64) / 8, 256, 0, stream>>>(
                      hb, Wp + (size_t)l * 16 * 32 * 32 * 16, hW);
        k_selfloop<<<(N_NODES * DIM) / 256, 256, 0, stream>>>(hW, dinv, bg + l * DIM, agg);
        k_edges   <<<(N_EDGES * 128) / 256, 256, 0, stream>>>(ei, dinv, hW, agg);
        k_update  <<<(N_NODES * DIM) / 256, 256, 0, stream>>>(agg, h, hb);
    }
    k_pool<<<(N_NODES * 128) / 256, 256, 0, stream>>>(h, batch, out);
}
